// Decoder_89309549953262
// MI455X (gfx1250) — matrix operand order repaired and checked by reading, not compiled, not run on hardware
//
#include <hip/hip_runtime.h>
#include <hip/hip_bf16.h>
#include <math.h>

// Problem constants (match reference)
constexpr int V = 50000, E = 512, H = 1024, B = 64, LQ = 32, LD = 400;

typedef __attribute__((ext_vector_type(16))) __bf16          v16bf;
typedef __attribute__((ext_vector_type(2)))  __bf16          v2bf;
typedef __attribute__((ext_vector_type(8)))  float           v8f;
typedef __attribute__((ext_vector_type(8)))  unsigned short  v8us;
typedef __attribute__((ext_vector_type(4)))  unsigned short  v4us;

// fp32x4 -> bf16x4 into fragment slots [o..o+3]
#if __has_builtin(__builtin_amdgcn_cvt_pk_bf16_f32)
__device__ inline void pack4(v16bf& v, int o, float4 f) {
  v2bf p0 = __builtin_amdgcn_cvt_pk_bf16_f32(f.x, f.y);
  v2bf p1 = __builtin_amdgcn_cvt_pk_bf16_f32(f.z, f.w);
  v[o + 0] = p0[0]; v[o + 1] = p0[1]; v[o + 2] = p1[0]; v[o + 3] = p1[1];
}
#else
__device__ inline void pack4(v16bf& v, int o, float4 f) {
  v[o + 0] = (__bf16)f.x; v[o + 1] = (__bf16)f.y;
  v[o + 2] = (__bf16)f.z; v[o + 3] = (__bf16)f.w;
}
#endif

// Fragment loaders.
// A 16-bit 16x32 fragment: lane m=l&15; element j<8 -> k=j+8*hi, j>=8 -> k=j+8+8*hi
// B 16-bit 32x16 fragment: lane n=l&15; element j<8 -> k=j+8*hi, j>=8 -> k=j+8+8*hi
template <typename T> struct Frag;

template <> struct Frag<float> {          // fp32 source: load b128 + packed cvt
  static __device__ inline v16bf a(const float* __restrict__ row, int off) {
    const float4* p = reinterpret_cast<const float4*>(row + off);
    v16bf v;
    pack4(v, 0, p[0]); pack4(v, 4, p[1]); pack4(v, 8, p[4]); pack4(v, 12, p[5]);
    return v;
  }
  static __device__ inline v16bf b(const float* __restrict__ row, int off) {
    const float4* p = reinterpret_cast<const float4*>(row + off);
    v16bf v;
    pack4(v, 0, p[0]); pack4(v, 4, p[1]); pack4(v, 8, p[4]); pack4(v, 12, p[5]);
    return v;
  }
};

template <> struct Frag<unsigned short> { // pre-converted bf16: pure b128 loads
  static __device__ inline v16bf a(const unsigned short* __restrict__ row, int off) {
    const v8us* p = reinterpret_cast<const v8us*>(row + off);
    union { v8us u[2]; v16bf b; } t;
    t.u[0] = p[0];      // k run [8*hi, +8)
    t.u[1] = p[2];      // k run [16+8*hi, +8)
    return t.b;
  }
  static __device__ inline v16bf b(const unsigned short* __restrict__ row, int off) {
    const v8us* p = reinterpret_cast<const v8us*>(row + off);
    union { v8us u[2]; v16bf b; } t;
    t.u[0] = p[0]; t.u[1] = p[2];
    return t.b;
  }
};

enum { EP_STORE = 0, EP_ATTN = 1 };

// D[m,n] = sum_k A[m,k] * W[n,k]   (A @ W^T), bf16 WMMA core.
// Each wave computes a 64(M) x 16(N) tile: 4 A-frags share 1 B-frag -> 4 WMMA/K-step.
// Block = 128 threads = 4 waves spread across N.  Requires M % 64 == 0, N % 16 == 0.
// mode EP_STORE: C[m,n] = D + bias[n]
// mode EP_ATTN : score[m] += sum_n tanh(D[m,n] + rowBias[m % Bmod][n]) * v[n]
template <typename T>
__global__ __launch_bounds__(128)
void wmma_gemm_bt(const T* __restrict__ A, int lda,
                  const T* __restrict__ W, int ldw,
                  float* __restrict__ C, int ldc,
                  const float* __restrict__ bias,
                  const float* __restrict__ rowBias,
                  const float* __restrict__ vvec,
                  float* __restrict__ score,
                  int M, int N, int K, int Bmod, int mode)
{
  const int wave = threadIdx.x >> 5;
  const int lane = threadIdx.x & 31;
  const int mbase = blockIdx.x * 64;
  const int nbase = (blockIdx.y * 4 + wave) * 16;
  if (nbase >= N) return;               // wave-uniform: EXEC stays all-ones for WMMA
  const int nloc = lane & 15, hi = lane >> 4;

  const T* Ar[4];
  Ar[0] = A + (size_t)(mbase + nloc) * lda;
  Ar[1] = Ar[0] + (size_t)16 * lda;
  Ar[2] = Ar[1] + (size_t)16 * lda;
  Ar[3] = Ar[2] + (size_t)16 * lda;
  const T* Wr = W + (size_t)(nbase + nloc) * ldw;

  v8f acc[4];
  #pragma unroll
  for (int f = 0; f < 4; ++f) acc[f] = (v8f){0.f,0.f,0.f,0.f,0.f,0.f,0.f,0.f};

  const int aoff = 8 * hi, boff = 8 * hi;
  for (int kb = 0; kb < K; kb += 32) {
    __builtin_prefetch(Wr + kb + 256, 0, 0);          // global_prefetch_b8
    v16bf bfrag = Frag<T>::b(Wr, kb + boff);          // shared by all 4 WMMAs
    v16bf a0 = Frag<T>::a(Ar[0], kb + aoff);
    v16bf a1 = Frag<T>::a(Ar[1], kb + aoff);
    v16bf a2 = Frag<T>::a(Ar[2], kb + aoff);
    v16bf a3 = Frag<T>::a(Ar[3], kb + aoff);
    acc[0] = __builtin_amdgcn_wmma_f32_16x16x32_bf16(false, a0, false, bfrag, (short)0, acc[0], false, false);
    acc[1] = __builtin_amdgcn_wmma_f32_16x16x32_bf16(false, a1, false, bfrag, (short)0, acc[1], false, false);
    acc[2] = __builtin_amdgcn_wmma_f32_16x16x32_bf16(false, a2, false, bfrag, (short)0, acc[2], false, false);
    acc[3] = __builtin_amdgcn_wmma_f32_16x16x32_bf16(false, a3, false, bfrag, (short)0, acc[3], false, false);
  }

  const int n = nbase + nloc;
  if (mode == EP_ATTN) {
    const float vv = vvec[n];
    #pragma unroll
    for (int f = 0; f < 4; ++f) {
      #pragma unroll
      for (int r = 0; r < 8; ++r) {
        int m = mbase + 16 * f + r + 8 * hi;   // D layout: VGPR r = row r (+8 in hi half)
        float val = tanhf(acc[f][r] + rowBias[(size_t)(m % Bmod) * N + n]) * vv;
        #pragma unroll
        for (int off = 8; off >= 1; off >>= 1) // reduce over 16-lane n-group
          val += __shfl_xor(val, off, 32);
        if (nloc == 0) atomicAdd(&score[m], val);   // global_atomic_add_f32
      }
    }
  } else {
    const float bv = bias ? bias[n] : 0.f;
    #pragma unroll
    for (int f = 0; f < 4; ++f) {
      #pragma unroll
      for (int r = 0; r < 8; ++r) {
        int m = mbase + 16 * f + r + 8 * hi;
        C[(size_t)m * ldc + n] = acc[f][r] + bv;
      }
    }
  }
}

// streaming fp32 -> bf16 (4 elems/thread); n must be a multiple of 4
__global__ void k_f2bf(const float4* __restrict__ in, v4us* __restrict__ out, long n4) {
  long i = blockIdx.x * (long)blockDim.x + threadIdx.x;
  if (i >= n4) return;
  float4 f = in[i];
#if __has_builtin(__builtin_amdgcn_cvt_pk_bf16_f32)
  union { v4us u; v2bf p[2]; } t;
  t.p[0] = __builtin_amdgcn_cvt_pk_bf16_f32(f.x, f.y);
  t.p[1] = __builtin_amdgcn_cvt_pk_bf16_f32(f.z, f.w);
#else
  union { v4us u; __bf16 b[4]; } t;
  t.b[0] = (__bf16)f.x; t.b[1] = (__bf16)f.y; t.b[2] = (__bf16)f.z; t.b[3] = (__bf16)f.w;
#endif
  out[i] = t.u;
}

// ---------------- small wave32-aware kernels ----------------

template <bool MAXRED>
__device__ inline float block_red(float v, float* red) {
  #pragma unroll
  for (int o = 16; o >= 1; o >>= 1) {
    float t = __shfl_xor(v, o, 32);
    v = MAXRED ? fmaxf(v, t) : (v + t);
  }
  int wid = threadIdx.x >> 5, lane = threadIdx.x & 31;
  if (lane == 0) red[wid] = v;
  __syncthreads();
  int nw = blockDim.x >> 5;
  if (wid == 0) {
    float x = (lane < nw) ? red[lane] : (MAXRED ? -1e30f : 0.f);
    #pragma unroll
    for (int o = 16; o >= 1; o >>= 1) {
      float t = __shfl_xor(x, o, 32);
      x = MAXRED ? fmaxf(x, t) : (x + t);
    }
    if (lane == 0) red[0] = x;
  }
  __syncthreads();
  float r = red[0];
  __syncthreads();
  return r;
}

__global__ void k_embed(const int* __restrict__ idx, const float* __restrict__ emb,
                        float* __restrict__ out) {
  int b = blockIdx.x;
  int v = idx[b];
  for (int e = threadIdx.x; e < E; e += blockDim.x)
    out[(size_t)b * E + e] = emb[(size_t)v * E + e];
}

__global__ void k_concat2(float* __restrict__ dst, const float* __restrict__ s1, int w1,
                          const float* __restrict__ s2, int w2) {
  int b = blockIdx.x;
  int w = w1 + w2;
  for (int i = threadIdx.x; i < w; i += blockDim.x)
    dst[(size_t)b * w + i] = (i < w1) ? s1[(size_t)b * w1 + i]
                                      : s2[(size_t)b * w2 + (i - w1)];
}

// score: [T, B]; enc: [T, B, H]; ctx: [B, H]; wout (opt): [B, T]
__global__ void k_attn_softmax_ctx(const float* __restrict__ score,
                                   const float* __restrict__ enc,
                                   float* __restrict__ ctx,
                                   float* __restrict__ wout, int T) {
  __shared__ float sw[416];
  __shared__ float red[8];
  int b = blockIdx.x;

  float mx = -1e30f;
  for (int t = threadIdx.x; t < T; t += blockDim.x) {
    float s = score[(size_t)t * B + b];
    sw[t] = s;
    mx = fmaxf(mx, s);
  }
  mx = block_red<true>(mx, red);

  float sum = 0.f;
  for (int t = threadIdx.x; t < T; t += blockDim.x) {
    float e = __expf(sw[t] - mx);
    sw[t] = e;
    sum += e;
  }
  sum = block_red<false>(sum, red);
  float inv = 1.f / sum;

  for (int t = threadIdx.x; t < T; t += blockDim.x) {
    float w = sw[t] * inv;
    sw[t] = w;
    if (wout) wout[(size_t)b * T + t] = w;
  }
  __syncthreads();

  for (int h = threadIdx.x; h < H; h += blockDim.x) {
    float acc = 0.f;
    for (int t = 0; t < T; ++t)
      acc += sw[t] * enc[((size_t)t * B + b) * H + h];
    ctx[(size_t)b * H + h] = acc;
  }
}

__device__ inline float sigm(float x) { return 1.f / (1.f + __expf(-x)); }

// gates = g1 + g2 + bi + bh; standard LSTM cell
__global__ void k_lstm(const float* __restrict__ g1, const float* __restrict__ g2,
                       const float* __restrict__ bi, const float* __restrict__ bh,
                       const float* __restrict__ cprev,
                       float* __restrict__ hout, float* __restrict__ cout) {
  int b = blockIdx.x;
  for (int h = threadIdx.x; h < H; h += blockDim.x) {
    size_t base = (size_t)b * 4 * H + h;
    float gi = sigm (g1[base + 0*H] + g2[base + 0*H] + bi[h + 0*H] + bh[h + 0*H]);
    float gf = sigm (g1[base + 1*H] + g2[base + 1*H] + bi[h + 1*H] + bh[h + 1*H]);
    float gg = tanhf(g1[base + 2*H] + g2[base + 2*H] + bi[h + 2*H] + bh[h + 2*H]);
    float go = sigm (g1[base + 3*H] + g2[base + 3*H] + bi[h + 3*H] + bh[h + 3*H]);
    float c  = gf * cprev[(size_t)b * H + h] + gi * gg;
    cout[(size_t)b * H + h] = c;
    hout[(size_t)b * H + h] = go * tanhf(c);
  }
}

__global__ void k_logsoftmax(float* __restrict__ out) {
  __shared__ float red[8];
  int b = blockIdx.x;
  float* row = out + (size_t)b * V;
  float mx = -1e30f;
  for (int i = threadIdx.x; i < V; i += blockDim.x) mx = fmaxf(mx, row[i]);
  mx = block_red<true>(mx, red);
  float s = 0.f;
  for (int i = threadIdx.x; i < V; i += blockDim.x) s += __expf(row[i] - mx);
  s = block_red<false>(s, red);
  float lse = mx + __logf(s);
  for (int i = threadIdx.x; i < V; i += blockDim.x) row[i] -= lse;
}

// ---------------- host side ----------------

template <typename T>
static inline void launch_gemm(hipStream_t s,
                               const T* A, int lda, const T* W, int ldw,
                               float* C, int ldc, const float* bias,
                               const float* rowBias, const float* vvec, float* score,
                               int M, int N, int K, int Bmod, int mode) {
  dim3 grid(M / 64, (N + 63) / 64);   // wave tile 64x16, 4 waves/block across N
  wmma_gemm_bt<T><<<grid, dim3(128), 0, s>>>(A, lda, W, ldw, C, ldc, bias,
                                             rowBias, vvec, score, M, N, K, Bmod, mode);
}

static inline void conv_bf16(hipStream_t s, const float* in, unsigned short* out, long n) {
  long n4 = n / 4;
  k_f2bf<<<dim3((unsigned)((n4 + 255) / 256)), dim3(256), 0, s>>>(
      (const float4*)in, (v4us*)out, n4);
}

extern "C" void kernel_launch(void* const* d_in, const int* in_sizes, int n_in,
                              void* d_out, int out_size, void* d_ws, size_t ws_size,
                              hipStream_t stream) {
  (void)in_sizes; (void)n_in; (void)out_size; (void)ws_size;

  const int*   tok    = (const int*)  d_in[0];
  const float* h0     = (const float*)d_in[1];   // [B,H]
  const float* c0     = (const float*)d_in[2];
  const float* dh0    = (const float*)d_in[3];
  const float* dc0    = (const float*)d_in[4];
  const float* q_enc  = (const float*)d_in[5];   // [LQ,B,H]
  const float* d_enc  = (const float*)d_in[6];   // [LD,B,H]
  const float* embed  = (const float*)d_in[7];   // [V,E]
  const float* W_qa   = (const float*)d_in[8];   // [H,2H]
  const float* b_qa   = (const float*)d_in[9];
  const float* v_q    = (const float*)d_in[10];
  const float* W_da   = (const float*)d_in[11];  // [H,3H]
  const float* b_da   = (const float*)d_in[12];
  const float* v_d    = (const float*)d_in[13];
  const float* W_ih   = (const float*)d_in[14];  // [4H,H+E]
  const float* W_hh   = (const float*)d_in[15];  // [4H,H]
  const float* b_ih   = (const float*)d_in[16];
  const float* b_hh   = (const float*)d_in[17];
  const float* Wd_ih  = (const float*)d_in[18];  // [4H,H]
  const float* Wd_hh  = (const float*)d_in[19];  // [4H,H]
  const float* bd_ih  = (const float*)d_in[20];
  const float* bd_hh  = (const float*)d_in[21];
  const float* W_out  = (const float*)d_in[22];  // [V,2H]
  const float* b_out  = (const float*)d_in[23];

  float* o_out  = (float*)d_out;                 // [B,V]
  float* o_h1   = o_out  + (size_t)B * V;        // [B,H]
  float* o_c1   = o_h1   + (size_t)B * H;
  float* o_dh   = o_c1   + (size_t)B * H;
  float* o_dc   = o_dh   + (size_t)B * H;
  float* o_docw = o_dc   + (size_t)B * H;        // [B,LD]

  // fp32 workspace
  float* ws = (float*)d_ws;
  float* embedded = ws; ws += (size_t)B * E;
  float* A_hq     = ws; ws += (size_t)B * H;
  float* score_q  = ws; ws += (size_t)LQ * B;
  float* q_ctx    = ws; ws += (size_t)B * H;
  float* doc_q    = ws; ws += (size_t)B * 2 * H;
  float* A_hd     = ws; ws += (size_t)B * H;
  float* score_d  = ws; ws += (size_t)LD * B;
  float* doc_ctx  = ws; ws += (size_t)B * H;
  float* dg1      = ws; ws += (size_t)B * 4 * H;
  float* dg2      = ws; ws += (size_t)B * 4 * H;
  float* rnn_in   = ws; ws += (size_t)B * (H + E);
  float* g1       = ws; ws += (size_t)B * 4 * H;
  float* g2       = ws; ws += (size_t)B * 4 * H;
  float* cat2     = ws; ws += (size_t)B * 2 * H;

  // bf16 workspace (unsigned short storage), 16B-aligned by construction
  unsigned short* bw = (unsigned short*)ws;
  unsigned short* h0_b     = bw; bw += (size_t)B * H;
  unsigned short* dh0_b    = bw; bw += (size_t)B * H;
  unsigned short* qenc_b   = bw; bw += (size_t)LQ * B * H;
  unsigned short* denc_b   = bw; bw += (size_t)LD * B * H;
  unsigned short* Wqa_b    = bw; bw += (size_t)H * 2 * H;
  unsigned short* Wda_b    = bw; bw += (size_t)H * 3 * H;
  unsigned short* Wih_b    = bw; bw += (size_t)4 * H * (H + E);
  unsigned short* Whh_b    = bw; bw += (size_t)4 * H * H;
  unsigned short* Wdih_b   = bw; bw += (size_t)4 * H * H;
  unsigned short* Wdhh_b   = bw; bw += (size_t)4 * H * H;
  unsigned short* docq_b   = bw; bw += (size_t)B * 2 * H;
  unsigned short* docctx_b = bw; bw += (size_t)B * H;
  unsigned short* rnnin_b  = bw; bw += (size_t)B * (H + E);

  hipMemsetAsync(score_q, 0, (size_t)LQ * B * sizeof(float), stream);
  hipMemsetAsync(score_d, 0, (size_t)LD * B * sizeof(float), stream);

  // one-shot fp32->bf16 conversions (read once, convert once; removes all
  // redundant per-tile cvt work from the compute-bound GEMMs)
  conv_bf16(stream, h0,    h0_b,   (long)B * H);
  conv_bf16(stream, dh0,   dh0_b,  (long)B * H);
  conv_bf16(stream, q_enc, qenc_b, (long)LQ * B * H);
  conv_bf16(stream, d_enc, denc_b, (long)LD * B * H);
  conv_bf16(stream, W_qa,  Wqa_b,  (long)H * 2 * H);
  conv_bf16(stream, W_da,  Wda_b,  (long)H * 3 * H);
  conv_bf16(stream, W_ih,  Wih_b,  (long)4 * H * (H + E));
  conv_bf16(stream, W_hh,  Whh_b,  (long)4 * H * H);
  conv_bf16(stream, Wd_ih, Wdih_b, (long)4 * H * H);
  conv_bf16(stream, Wd_hh, Wdhh_b, (long)4 * H * H);

  // 0) embedding gather
  hipLaunchKernelGGL(k_embed, dim3(B), dim3(128), 0, stream, tok, embed, embedded);

  // 1) query attention
  // A_hq = h_last @ W_qa[:, :H].T + b_qa
  launch_gemm(stream, h0_b, H, Wqa_b, 2 * H, A_hq, H, b_qa,
              nullptr, nullptr, nullptr, B, H, H, B, EP_STORE);
  // score_q[t*B+b] = sum_n tanh(enc@Wqa[:,H:].T + A_hq[b]) * v_q   (fused)
  launch_gemm(stream, qenc_b, H, Wqa_b + H, 2 * H, (float*)nullptr, 0, nullptr,
              A_hq, v_q, score_q, LQ * B, H, H, B, EP_ATTN);
  hipLaunchKernelGGL(k_attn_softmax_ctx, dim3(B), dim3(256), 0, stream,
                     score_q, q_enc, q_ctx, (float*)nullptr, LQ);

  // 2) doc attention
  hipLaunchKernelGGL(k_concat2, dim3(B), dim3(256), 0, stream, doc_q, h0, H, q_ctx, H);
  conv_bf16(stream, doc_q, docq_b, (long)B * 2 * H);
  launch_gemm(stream, docq_b, 2 * H, Wda_b, 3 * H, A_hd, H, b_da,
              nullptr, nullptr, nullptr, B, H, 2 * H, B, EP_STORE);
  launch_gemm(stream, denc_b, H, Wda_b + 2 * H, 3 * H, (float*)nullptr, 0, nullptr,
              A_hd, v_d, score_d, LD * B, H, H, B, EP_ATTN);
  hipLaunchKernelGGL(k_attn_softmax_ctx, dim3(B), dim3(256), 0, stream,
                     score_d, d_enc, doc_ctx, o_docw, LD);
  conv_bf16(stream, doc_ctx, docctx_b, (long)B * H);

  // 3) distraction LSTM cell
  launch_gemm(stream, docctx_b, H, Wdih_b, H, dg1, 4 * H, nullptr,
              nullptr, nullptr, nullptr, B, 4 * H, H, B, EP_STORE);
  launch_gemm(stream, dh0_b, H, Wdhh_b, H, dg2, 4 * H, nullptr,
              nullptr, nullptr, nullptr, B, 4 * H, H, B, EP_STORE);
  hipLaunchKernelGGL(k_lstm, dim3(B), dim3(256), 0, stream,
                     dg1, dg2, bd_ih, bd_hh, dc0, o_dh, o_dc);

  // 4) main LSTM cell
  hipLaunchKernelGGL(k_concat2, dim3(B), dim3(256), 0, stream, rnn_in, embedded, E, doc_ctx, H);
  conv_bf16(stream, rnn_in, rnnin_b, (long)B * (H + E));
  launch_gemm(stream, rnnin_b, H + E, Wih_b, H + E, g1, 4 * H, nullptr,
              nullptr, nullptr, nullptr, B, 4 * H, H + E, B, EP_STORE);
  launch_gemm(stream, h0_b, H, Whh_b, H, g2, 4 * H, nullptr,
              nullptr, nullptr, nullptr, B, 4 * H, H, B, EP_STORE);
  hipLaunchKernelGGL(k_lstm, dim3(B), dim3(256), 0, stream,
                     g1, g2, b_ih, b_hh, c0, o_h1, o_c1);

  // 5) output projection + log_softmax.
  //    Stays fp32-in (in-kernel cvt): this GEMM is HBM-bound (410 MB of W_out,
  //    each element read exactly once) so pre-conversion would ADD traffic.
  hipLaunchKernelGGL(k_concat2, dim3(B), dim3(256), 0, stream, cat2, o_h1, H, doc_ctx, H);
  launch_gemm(stream, cat2, 2 * H, W_out, 2 * H, o_out, V, b_out,
              nullptr, nullptr, nullptr, B, V, 2 * H, B, EP_STORE);
  hipLaunchKernelGGL(k_logsoftmax, dim3(B), dim3(256), 0, stream, o_out);
}
